// ImplicitAttentionFields_81733227643075
// MI455X (gfx1250) — matrix operand order repaired and checked by reading, not compiled, not run on hardware
//
#include <hip/hip_runtime.h>
#include <hip/hip_fp16.h>
#include <math.h>

typedef __attribute__((ext_vector_type(16))) _Float16 v16h;
typedef __attribute__((ext_vector_type(8)))  _Float16 v8h;
typedef __attribute__((ext_vector_type(8)))  float    v8f;

#define NPTS  5000
#define BATCH 4
#define MA    16
#define KK    512
#define CDIM  128
#define QTOT  65536

// ---------------- workspace layout (byte offsets, 256-aligned) ----------------
#define WS_HW    0          // 90112 f16  : packed fragment-order f16 weights
#define WS_ANC   184320     // 192 f32    : anchors [4][16][3]
#define WS_IDX   185344     // 32768 i32  : knn indices [4][16][512]
#define WS_FEAT  316416     // 8192 f32   : pooled features [4][16][128]
#define WS_LAT   349184     // 8192 f32   : encoder latent [4][16][128]
#define WS_KV    381952     // 16384 f32  : cross K/V [4][2][16][128]

__device__ __forceinline__ float gelu_exact(float x) {
  return 0.5f * x * (1.0f + erff(x * 0.70710678118654752f));
}

__device__ __forceinline__ void atomicMaxF(float* addr, float val) {
  unsigned int* ua = (unsigned int*)addr;
  unsigned int old = *ua;
  while (__uint_as_float(old) < val) {
    unsigned int assumed = old;
    old = atomicCAS(ua, assumed, __float_as_uint(val));
    if (old == assumed) break;
  }
}

// ==================== weight f32 -> f16 packing (fragment-native order) ====================
// Packed halves layout per region: blocks of 512 halves = one 32x16 B-fragment,
// ordered [blk = nt*ksteps + ks][lane][e] so frag_b is one v16h (32B) load per lane.
// Region offsets (halves): 0 Wqproj(pad 64x128, ksteps=2) ; 8192 crossWq ; 24576 crossWo ;
// 40960 mlp0 ; 57344 mlp1 ; 73728 mlp2 (all ksteps=4)
__global__ void k_convw(const float* __restrict__ Wqp, const float* __restrict__ Wq,
                        const float* __restrict__ Wo,  const float* __restrict__ M0,
                        const float* __restrict__ M1,  const float* __restrict__ M2,
                        _Float16* __restrict__ hw) {
  int i = blockIdx.x * 256 + threadIdx.x;
  if (i >= 90112) return;
  int il, ksteps;
  const float* src;
  if (i < 8192)       { il = i;          ksteps = 2; src = Wqp; }
  else if (i < 24576) { il = i - 8192;   ksteps = 4; src = Wq;  }
  else if (i < 40960) { il = i - 24576;  ksteps = 4; src = Wo;  }
  else if (i < 57344) { il = i - 40960;  ksteps = 4; src = M0;  }
  else if (i < 73728) { il = i - 57344;  ksteps = 4; src = M1;  }
  else                { il = i - 73728;  ksteps = 4; src = M2;  }
  int e    = il & 15;
  int lane = (il >> 4) & 31;
  int blk  = il >> 9;
  int nt = blk / ksteps, ks = blk - nt * ksteps;
  int n = (nt << 4) + (lane & 15);
  int k = (ks << 5) + ((e >> 3) << 4) + ((lane >> 4) << 3) + (e & 7);
  float v;
  if (i < 8192) v = (k < 39) ? src[k * 128 + n] : 0.0f;   // zero-pad posenc rows 39..63
  else          v = src[k * 128 + n];
  hw[i] = (_Float16)v;
}

__global__ void k_initfeat(float* feat) {
  int i = blockIdx.x * 256 + threadIdx.x;
  if (i < BATCH * MA * CDIM) feat[i] = -3.0e38f;
}

// ==================== farthest point sampling ====================
__global__ void k_fps(const float* __restrict__ pts, float* __restrict__ anchors) {
  extern __shared__ char dsm[];
  float* dist = (float*)dsm;          // NPTS
  float* rv   = dist + NPTS;          // 256
  int*   ri   = (int*)(rv + 256);     // 256
  __shared__ float cur[3];
  __shared__ int bsel;
  int b = blockIdx.x, t = threadIdx.x;
  const float* P = pts + (size_t)b * NPTS * 3;
  float* A = anchors + b * MA * 3;
  if (t < 3) { cur[t] = P[t]; A[t] = P[t]; }
  __syncthreads();
  for (int i = t; i < NPTS; i += blockDim.x) {
    float dx = P[3*i] - cur[0], dy = P[3*i+1] - cur[1], dz = P[3*i+2] - cur[2];
    dist[i] = dx*dx + dy*dy + dz*dz;
  }
  __syncthreads();
  for (int s = 1; s < MA; ++s) {
    float bv = -1.0f; int bi = 0;
    for (int i = t; i < NPTS; i += blockDim.x)
      if (dist[i] > bv) { bv = dist[i]; bi = i; }
    rv[t] = bv; ri[t] = bi;
    __syncthreads();
    for (int off = 128; off > 0; off >>= 1) {
      if (t < off && rv[t + off] > rv[t]) { rv[t] = rv[t + off]; ri[t] = ri[t + off]; }
      __syncthreads();
    }
    if (t == 0) bsel = ri[0];
    __syncthreads();
    if (t < 3) { cur[t] = P[3 * bsel + t]; A[s * 3 + t] = cur[t]; }
    __syncthreads();
    for (int i = t; i < NPTS; i += blockDim.x) {
      float dx = P[3*i] - cur[0], dy = P[3*i+1] - cur[1], dz = P[3*i+2] - cur[2];
      float d = dx*dx + dy*dy + dz*dz;
      if (d < dist[i]) dist[i] = d;
    }
    __syncthreads();
  }
}

// ==================== KNN-512 select (bitwise threshold search) ====================
__global__ void k_knn(const float* __restrict__ pts, const float* __restrict__ anchors,
                      int* __restrict__ idxsel) {
  extern __shared__ char dsm[];
  unsigned* db = (unsigned*)dsm;      // NPTS squared-distance bit patterns (>=0 floats)
  __shared__ int csum;
  __shared__ int cnt;
  int bm = blockIdx.x, b = bm >> 4, t = threadIdx.x;
  const float* P = pts + (size_t)b * NPTS * 3;
  float ax = anchors[bm*3], ay = anchors[bm*3+1], az = anchors[bm*3+2];
  for (int i = t; i < NPTS; i += blockDim.x) {
    float dx = P[3*i] - ax, dy = P[3*i+1] - ay, dz = P[3*i+2] - az;
    db[i] = __float_as_uint(dx*dx + dy*dy + dz*dz);
  }
  __syncthreads();
  unsigned lo = 0u, hi = 0xFFFFFFFFu;
  for (int it = 0; it < 32; ++it) {
    unsigned mid = lo + ((hi - lo) >> 1);
    if (t == 0) csum = 0;
    __syncthreads();
    int c = 0;
    for (int i = t; i < NPTS; i += blockDim.x) c += (db[i] <= mid) ? 1 : 0;
    atomicAdd(&csum, c);
    __syncthreads();
    int tot = csum;
    __syncthreads();
    if (tot >= KK) hi = mid; else lo = mid + 1;
  }
  if (t == 0) cnt = 0;
  __syncthreads();
  for (int i = t; i < NPTS; i += blockDim.x) {
    if (db[i] <= lo) {
      int p = atomicAdd(&cnt, 1);
      if (p < KK) idxsel[bm * KK + p] = i;
    }
  }
}

// ==================== mini-PointNet + max pool ====================
__global__ void k_pointnet(const float* __restrict__ pts, const int* __restrict__ idxsel,
                           const float* __restrict__ W1, const float* __restrict__ b1,
                           const float* __restrict__ W2, const float* __restrict__ b2,
                           const float* __restrict__ W3, const float* __restrict__ b3,
                           float* __restrict__ feat) {
  int g = blockIdx.x * 256 + threadIdx.x;  // one thread per (b,m,k) point
  int bm = g >> 9;
  int b = bm >> 4;
  int pi = idxsel[g];
  const float* p = pts + ((size_t)b * NPTS + pi) * 3;
  float x0 = p[0], x1 = p[1], x2 = p[2];
  float h1[64];
  for (int j = 0; j < 64; ++j) {
    float v = b1[j] + x0 * W1[j] + x1 * W1[64 + j] + x2 * W1[128 + j];
    h1[j] = v > 0.0f ? v : 0.0f;
  }
  float acc[CDIM];
  for (int c = 0; c < CDIM; ++c) acc[c] = b3[c];
  for (int j = 0; j < CDIM; ++j) {
    float h2 = b2[j];
    for (int i = 0; i < 64; ++i) h2 += h1[i] * W2[i * CDIM + j];
    if (h2 > 0.0f)
      for (int c = 0; c < CDIM; ++c) acc[c] += h2 * W3[j * CDIM + c];
  }
  for (int c = 0; c < CDIM; ++c) atomicMaxF(&feat[bm * CDIM + c], acc[c]);
}

// ==================== encoder transformer (tiny: 16 tokens/batch, f32 VALU) ====================
struct EncBlkP {
  const float *Wq,*Wk,*Wv,*bq,*bk,*bv,*Wo,*bo,*g1,*be1,*Wf1,*bf1,*Wf2,*bf2,*g2,*be2;
};
struct EncP {
  const float *feat, *anchors, *apW, *apb;
  EncBlkP blk[4];
  float* latent;
};

__global__ void k_encoder(EncP p) {
  extern __shared__ char dsm[];
  float* x  = (float*)dsm;     // 2048
  float* q  = x + 2048;        // 2048
  float* k  = q + 2048;        // 2048
  float* v  = k + 2048;        // 2048
  float* o  = v + 2048;        // 2048
  float* f  = o + 2048;        // 4096 (16x256)
  float* pe = f + 4096;        // 640 (16x40)
  __shared__ float rmu[16], rvr[16];
  int b = blockIdx.x, t = threadIdx.x;
  for (int i = t; i < MA * 39; i += blockDim.x) {
    int m = i / 39, j = i % 39;
    const float* a = p.anchors + (b * MA + m) * 3;
    float val;
    if (j < 3) val = a[j];
    else {
      int r = j - 3, fr = r / 6, c = r % 6;
      float s = exp2f((float)fr) * a[c % 3];
      val = (c < 3) ? sinf(s) : cosf(s);
    }
    pe[m * 40 + j] = val;
  }
  __syncthreads();
  for (int i = t; i < MA * CDIM; i += blockDim.x) {
    int m = i >> 7, n = i & 127;
    float acc = p.apb[n] + p.feat[(b * MA + m) * CDIM + n];
    for (int j = 0; j < 39; ++j) acc += pe[m * 40 + j] * p.apW[j * CDIM + n];
    x[i] = acc;
  }
  __syncthreads();
  for (int blk = 0; blk < 4; ++blk) {
    EncBlkP B = p.blk[blk];
    for (int i = t; i < MA * CDIM; i += blockDim.x) {
      int m = i >> 7, n = i & 127;
      float aq = B.bq[n], ak = B.bk[n], av = B.bv[n];
      for (int j = 0; j < CDIM; ++j) {
        float xv = x[m * CDIM + j];
        aq += xv * B.Wq[j * CDIM + n];
        ak += xv * B.Wk[j * CDIM + n];
        av += xv * B.Wv[j * CDIM + n];
      }
      q[i] = aq; k[i] = ak; v[i] = av;
    }
    __syncthreads();
    if (t < 128) {
      int m = t >> 3, h = t & 7;
      float sc[16]; float mx = -3.0e38f;
      for (int j = 0; j < 16; ++j) {
        float s = 0.0f;
        for (int d = 0; d < 16; ++d) s += q[m*CDIM + h*16 + d] * k[j*CDIM + h*16 + d];
        s *= 0.25f;
        sc[j] = s; if (s > mx) mx = s;
      }
      float sum = 0.0f;
      for (int j = 0; j < 16; ++j) { sc[j] = expf(sc[j] - mx); sum += sc[j]; }
      float inv = 1.0f / sum;
      for (int d = 0; d < 16; ++d) {
        float a = 0.0f;
        for (int j = 0; j < 16; ++j) a += sc[j] * v[j*CDIM + h*16 + d];
        o[m*CDIM + h*16 + d] = a * inv;
      }
    }
    __syncthreads();
    for (int i = t; i < MA * CDIM; i += blockDim.x) {
      int m = i >> 7, n = i & 127;
      float acc = B.bo[n] + x[i];
      for (int j = 0; j < CDIM; ++j) acc += o[m * CDIM + j] * B.Wo[j * CDIM + n];
      q[i] = acc;
    }
    __syncthreads();
    if (t < 16) {
      float s = 0.0f, ss = 0.0f;
      for (int n = 0; n < CDIM; ++n) { float vv = q[t * CDIM + n]; s += vv; ss += vv * vv; }
      float mu = s / CDIM;
      rmu[t] = mu; rvr[t] = ss / CDIM - mu * mu;
    }
    __syncthreads();
    for (int i = t; i < MA * CDIM; i += blockDim.x) {
      int m = i >> 7, n = i & 127;
      x[i] = (q[i] - rmu[m]) * rsqrtf(rvr[m] + 1e-5f) * B.g1[n] + B.be1[n];
    }
    __syncthreads();
    for (int i = t; i < MA * 256; i += blockDim.x) {
      int m = i >> 8, n = i & 255;
      float acc = B.bf1[n];
      for (int j = 0; j < CDIM; ++j) acc += x[m * CDIM + j] * B.Wf1[j * 256 + n];
      f[i] = gelu_exact(acc);
    }
    __syncthreads();
    for (int i = t; i < MA * CDIM; i += blockDim.x) {
      int m = i >> 7, n = i & 127;
      float acc = B.bf2[n] + x[i];
      for (int j = 0; j < 256; ++j) acc += f[m * 256 + j] * B.Wf2[j * CDIM + n];
      q[i] = acc;
    }
    __syncthreads();
    if (t < 16) {
      float s = 0.0f, ss = 0.0f;
      for (int n = 0; n < CDIM; ++n) { float vv = q[t * CDIM + n]; s += vv; ss += vv * vv; }
      float mu = s / CDIM;
      rmu[t] = mu; rvr[t] = ss / CDIM - mu * mu;
    }
    __syncthreads();
    for (int i = t; i < MA * CDIM; i += blockDim.x) {
      int m = i >> 7, n = i & 127;
      x[i] = (q[i] - rmu[m]) * rsqrtf(rvr[m] + 1e-5f) * B.g2[n] + B.be2[n];
    }
    __syncthreads();
  }
  for (int i = t; i < MA * CDIM; i += blockDim.x) p.latent[b * MA * CDIM + i] = x[i];
}

// ==================== latent K/V projection ====================
__global__ void k_kv(const float* __restrict__ latent,
                     const float* __restrict__ Wk, const float* __restrict__ bk,
                     const float* __restrict__ Wv, const float* __restrict__ bv,
                     float* __restrict__ kv) {
  int b = blockIdx.x, t = threadIdx.x;
  for (int i = t; i < MA * CDIM; i += blockDim.x) {
    int m = i >> 7, n = i & 127;
    float ak = bk[n], av = bv[n];
    for (int j = 0; j < CDIM; ++j) {
      float l = latent[b * MA * CDIM + m * CDIM + j];
      ak += l * Wk[j * CDIM + n];
      av += l * Wv[j * CDIM + n];
    }
    kv[b * 2 * MA * CDIM + i] = ak;
    kv[b * 2 * MA * CDIM + MA * CDIM + i] = av;
  }
}

// ==================== WMMA decoder (16 queries per wave) ====================
// A fragment: two contiguous 8-half runs per lane (ISA 16-bit A layout) -> 2x b128 LDS loads
__device__ __forceinline__ v16h frag_a(const _Float16* As, int ld, int k0, int lane) {
  int m = lane & 15;
  int kh = (lane >> 4) << 3;
  const _Float16* base = As + m * ld + k0 + kh;
  v8h a0 = *(const v8h*)base;          // K = k0+kh .. +7
  v8h a1 = *(const v8h*)(base + 16);   // K = k0+16+kh .. +7
  v16h a;
#pragma unroll
  for (int e = 0; e < 8; ++e) { a[e] = a0[e]; a[8 + e] = a1[e]; }
  return a;
}

__device__ __forceinline__ void gemm_f32(const _Float16* As, int kdim,
                                         const _Float16* Wp, const float* bias,
                                         float* S, int lane) {
  const int ksteps = kdim >> 5;
  const v16h* Wv = (const v16h*)Wp;    // fragment-packed: one 32B load per lane per WMMA
  for (int nt = 0; nt < 8; ++nt) {
    v8f acc = {};
    for (int ks = 0; ks < ksteps; ++ks) {
      v16h a  = frag_a(As, kdim, ks << 5, lane);
      v16h bb = Wv[(nt * ksteps + ks) * 32 + lane];
      acc = __builtin_amdgcn_wmma_f32_16x16x32_f16(false, a, false, bb, (short)0, acc, false, false);
    }
    int n = (nt << 4) + (lane & 15);
    float bv = bias[n];
    int mb = (lane >> 4) << 3;
#pragma unroll
    for (int r = 0; r < 8; ++r) S[(mb + r) * CDIM + n] = acc[r] + bv;
  }
}

__device__ __forceinline__ void gemm_gelu_f16(const _Float16* As, const _Float16* Wp,
                                              const float* bias, _Float16* Out, int lane) {
  const v16h* Wv = (const v16h*)Wp;
  for (int nt = 0; nt < 8; ++nt) {
    v8f acc = {};
#pragma unroll
    for (int ks = 0; ks < 4; ++ks) {
      v16h a  = frag_a(As, CDIM, ks << 5, lane);
      v16h bb = Wv[(nt * 4 + ks) * 32 + lane];
      acc = __builtin_amdgcn_wmma_f32_16x16x32_f16(false, a, false, bb, (short)0, acc, false, false);
    }
    int n = (nt << 4) + (lane & 15);
    float bv = bias[n];
    int mb = (lane >> 4) << 3;
#pragma unroll
    for (int r = 0; r < 8; ++r)
      Out[(mb + r) * CDIM + n] = (_Float16)gelu_exact(acc[r] + bv);
  }
}

struct DecP {
  const float* qpts;
  const float* bqp;                    // q_proj.b
  const float* Wql; const float* bql;  // q_lin
  const float* bq;  const float* bo;   // cross biases
  const float* bm0; const float* bm1; const float* bm2;
  const float* Wlast; const float* blast;
  const float* kv;                     // [4][2][16][128] f32
  const _Float16* hw;                  // packed f16 weights
  float* out;
};

#define DEC_WAVES 2
#define DEC_WAVE_BYTES 16640           // xyz(256) + A0(4096) + A1(4096) + S(8192)

__global__ void k_decoder(DecP p) {
  extern __shared__ char dsm[];
  int lane = threadIdx.x & 31;
  int wv = threadIdx.x >> 5;
  float* kbuf = (float*)dsm;                          // [16][128]
  float* vbuf = kbuf + MA * CDIM;                     // [16][128]
  char* wb = dsm + 16384 + wv * DEC_WAVE_BYTES;
  float*    xyz = (float*)wb;                         // [16][4]
  _Float16* A0  = (_Float16*)(wb + 256);              // [16][128] f16
  _Float16* A1  = A0 + MA * CDIM;                     // [16][128] f16
  float*    S   = (float*)(wb + 256 + 8192);          // [16][128] f32

  int tile = blockIdx.x * DEC_WAVES + wv;             // 16384 tiles total
  int b = tile >> 12;                                 // 4096 tiles / batch
  int q0 = (tile & 4095) << 4;

  // warm L2/WGP$ with the packed weights (global_prefetch_b8, no counters)
  if (wv == 0) {
    for (int off = lane * 64; off < 90112; off += 2048)
      __builtin_prefetch((const void*)(p.hw + off), 0, 1);
  }

  const float* kvsrc = p.kv + (size_t)b * 2 * MA * CDIM;
  for (int i = threadIdx.x; i < 2 * MA * CDIM; i += blockDim.x) kbuf[i] = kvsrc[i];
  __syncthreads();

  const float* QP = p.qpts + ((size_t)b * QTOT + q0) * 3;
  if (lane < 16) {
    xyz[lane*4+0] = QP[lane*3+0];
    xyz[lane*4+1] = QP[lane*3+1];
    xyz[lane*4+2] = QP[lane*3+2];
    xyz[lane*4+3] = 0.0f;
  }
  // posenc -> A0 (16x64, zero-padded beyond 39)
  for (int i = lane; i < MA * 64; i += 32) {
    int m = i >> 6, kk2 = i & 63;
    float val = 0.0f;
    if (kk2 < 3) val = xyz[m*4 + kk2];
    else if (kk2 < 39) {
      int r = kk2 - 3, fr = r / 6, c = r % 6;
      float s = exp2f((float)fr) * xyz[m*4 + (c % 3)];
      val = (c < 3) ? sinf(s) : cosf(s);
    }
    A0[i] = (_Float16)val;
  }
  const _Float16* hWqp = p.hw;
  const _Float16* hWq  = p.hw + 8192;
  const _Float16* hWo  = p.hw + 24576;
  const _Float16* hM0  = p.hw + 40960;
  const _Float16* hM1  = p.hw + 57344;
  const _Float16* hM2  = p.hw + 73728;

  gemm_f32(A0, 64, hWqp, p.bqp, S, lane);                   // dq = posenc @ Wqproj + b
  for (int i = lane; i < MA * CDIM; i += 32) A1[i] = (_Float16)S[i];
  gemm_f32(A1, CDIM, hWq, p.bq, S, lane);                   // q = dq @ Wq + bq

  // fused 16-key cross-attention (scores/softmax/AV), writes attn-out f16 to A0
  for (int task = lane; task < 128; task += 32) {
    int m = task >> 3, h = task & 7;
    float sc[16]; float mx = -3.0e38f;
#pragma unroll
    for (int j = 0; j < 16; ++j) {
      float s = 0.0f;
#pragma unroll
      for (int d = 0; d < 16; ++d) s += S[m*CDIM + h*16 + d] * kbuf[j*CDIM + h*16 + d];
      s *= 0.25f;                                           // 1/sqrt(dh=16)
      sc[j] = s; if (s > mx) mx = s;
    }
    float sum = 0.0f;
#pragma unroll
    for (int j = 0; j < 16; ++j) { sc[j] = expf(sc[j] - mx); sum += sc[j]; }
    float inv = 1.0f / sum;
#pragma unroll
    for (int d = 0; d < 16; ++d) {
      float a = 0.0f;
#pragma unroll
      for (int j = 0; j < 16; ++j) a += sc[j] * vbuf[j*CDIM + h*16 + d];
      A0[m*CDIM + h*16 + d] = (_Float16)(a * inv);
    }
  }
  gemm_f32(A0, CDIM, hWo, p.bo, S, lane);                   // attn @ Wo + bo

  // out = 0.5*dql + 0.5*attn  (WEIGHT = 0.5); dql = xyz @ Wql + bql computed inline
  for (int i = lane; i < MA * CDIM; i += 32) {
    int m = i >> 7, n = i & 127;
    float d = p.bql[n]
            + xyz[m*4+0] * p.Wql[n]
            + xyz[m*4+1] * p.Wql[CDIM + n]
            + xyz[m*4+2] * p.Wql[2*CDIM + n];
    A1[i] = (_Float16)(0.5f * d + 0.5f * S[i]);
  }
  gemm_gelu_f16(A1, hM0, p.bm0, A0, lane);                  // MLP layer 0
  gemm_gelu_f16(A0, hM1, p.bm1, A1, lane);                  // MLP layer 1
  gemm_gelu_f16(A1, hM2, p.bm2, A0, lane);                  // MLP layer 2

  if (lane < 16) {                                          // final 128 -> 1
    float acc = p.blast[0];
    for (int n = 0; n < CDIM; ++n) acc += (float)A0[lane * CDIM + n] * p.Wlast[n];
    p.out[(size_t)b * QTOT + q0 + lane] = acc;
  }
}

// ==================== host launch ====================
extern "C" void kernel_launch(void* const* d_in, const int* in_sizes, int n_in,
                              void* d_out, int out_size, void* d_ws, size_t ws_size,
                              hipStream_t stream) {
  (void)in_sizes; (void)n_in; (void)out_size; (void)ws_size;
  const float* pc = (const float*)d_in[0];
  const float* qp = (const float*)d_in[1];
  char* ws = (char*)d_ws;
  _Float16* hw   = (_Float16*)(ws + WS_HW);
  float* anchors = (float*)(ws + WS_ANC);
  int*   idxsel  = (int*)(ws + WS_IDX);
  float* feat    = (float*)(ws + WS_FEAT);
  float* latent  = (float*)(ws + WS_LAT);
  float* kvb     = (float*)(ws + WS_KV);

  // input index map (setup_inputs dict order, depth-first):
  // 0 pc, 1 qp, 2..7 mini(W1,b1,W2,b2,W3,b3), 8..9 anchor_proj(W,b),
  // 10+16e enc blocks (Wq,Wk,Wv,bq,bk,bv,Wo,bo,ln1g,ln1b,Wf1,bf1,Wf2,bf2,ln2g,ln2b),
  // 74..75 q_proj, 76..77 q_lin, 78..85 cross(Wq,Wk,Wv,bq,bk,bv,Wo,bo),
  // 86..93 mlp(W,b)x4
  k_convw<<<352, 256, 0, stream>>>((const float*)d_in[74], (const float*)d_in[78],
                                   (const float*)d_in[84], (const float*)d_in[86],
                                   (const float*)d_in[88], (const float*)d_in[90], hw);
  k_initfeat<<<32, 256, 0, stream>>>(feat);
  k_fps<<<BATCH, 256, (NPTS + 512) * 4, stream>>>(pc, anchors);
  k_knn<<<BATCH * MA, 256, NPTS * 4, stream>>>(pc, anchors, idxsel);
  k_pointnet<<<(BATCH * MA * KK) / 256, 256, 0, stream>>>(
      pc, idxsel,
      (const float*)d_in[2], (const float*)d_in[3], (const float*)d_in[4],
      (const float*)d_in[5], (const float*)d_in[6], (const float*)d_in[7], feat);

  EncP ep;
  ep.feat = feat; ep.anchors = anchors;
  ep.apW = (const float*)d_in[8]; ep.apb = (const float*)d_in[9];
  for (int e = 0; e < 4; ++e) {
    int base = 10 + 16 * e;
    ep.blk[e].Wq  = (const float*)d_in[base + 0];
    ep.blk[e].Wk  = (const float*)d_in[base + 1];
    ep.blk[e].Wv  = (const float*)d_in[base + 2];
    ep.blk[e].bq  = (const float*)d_in[base + 3];
    ep.blk[e].bk  = (const float*)d_in[base + 4];
    ep.blk[e].bv  = (const float*)d_in[base + 5];
    ep.blk[e].Wo  = (const float*)d_in[base + 6];
    ep.blk[e].bo  = (const float*)d_in[base + 7];
    ep.blk[e].g1  = (const float*)d_in[base + 8];
    ep.blk[e].be1 = (const float*)d_in[base + 9];
    ep.blk[e].Wf1 = (const float*)d_in[base + 10];
    ep.blk[e].bf1 = (const float*)d_in[base + 11];
    ep.blk[e].Wf2 = (const float*)d_in[base + 12];
    ep.blk[e].bf2 = (const float*)d_in[base + 13];
    ep.blk[e].g2  = (const float*)d_in[base + 14];
    ep.blk[e].be2 = (const float*)d_in[base + 15];
  }
  ep.latent = latent;
  k_encoder<<<BATCH, 256, 14976 * 4, stream>>>(ep);
  k_kv<<<BATCH, 256, 0, stream>>>(latent, (const float*)d_in[79], (const float*)d_in[82],
                                  (const float*)d_in[80], (const float*)d_in[83], kvb);

  DecP dp;
  dp.qpts = qp;
  dp.bqp = (const float*)d_in[75];
  dp.Wql = (const float*)d_in[76]; dp.bql = (const float*)d_in[77];
  dp.bq  = (const float*)d_in[81]; dp.bo  = (const float*)d_in[85];
  dp.bm0 = (const float*)d_in[87]; dp.bm1 = (const float*)d_in[89]; dp.bm2 = (const float*)d_in[91];
  dp.Wlast = (const float*)d_in[92]; dp.blast = (const float*)d_in[93];
  dp.kv = kvb; dp.hw = hw; dp.out = (float*)d_out;
  // 16384 tiles / 2 waves-per-block; 49664B LDS (kv 16KB + 2 x 16640B per-wave)
  k_decoder<<<(BATCH * QTOT / 16) / DEC_WAVES, 32 * DEC_WAVES,
              16384 + DEC_WAVES * DEC_WAVE_BYTES, stream>>>(dp);
}
